// WindowAttention_22247930593458
// MI455X (gfx1250) — matrix operand order repaired and checked by reading, not compiled, not run on hardware
//
#include <hip/hip_runtime.h>

typedef __attribute__((ext_vector_type(16))) _Float16 v16h;
typedef __attribute__((ext_vector_type(8)))  _Float16 v8h;
typedef __attribute__((ext_vector_type(4)))  _Float16 v4h;
typedef __attribute__((ext_vector_type(8)))  float    v8f;
typedef __attribute__((ext_vector_type(4)))  unsigned int u32x4;
typedef __attribute__((ext_vector_type(8)))  int      i32x8;
typedef __attribute__((ext_vector_type(4)))  int      i32x4;

#define NTOK   49
#define NH_    12
#define HD_    32
#define C_     384
#define BW_    2048
#define NWIN   64
#define SCALE_ 0.17677669529663687f   // 32^-0.5

// ---------------------------------------------------------------------------
// WMMA fragment loaders (wave32); A and B both use contiguous k order k = 16*(lane/16) + s.
__device__ __forceinline__ v16h load_a16(const _Float16* base, int ld, int lane) {
  const int m  = lane & 15;
  const int kb = (lane & 16) ? 16 : 0;
  const _Float16* r = base + m * ld + kb;
  v16h a;
#pragma unroll
  for (int i = 0; i < 4; ++i) { a[2*i]   = r[2*i];    a[2*i+1] = r[2*i+1]; }
#pragma unroll
  for (int i = 0; i < 4; ++i) { a[8+2*i] = r[8+2*i]; a[9+2*i] = r[8+2*i+1]; }
  return a;
}
// B staged transposed in LDS as [N][K]; vgpr j holds contiguous K-pair.
__device__ __forceinline__ v16h load_b16(const _Float16* base, int ld, int lane) {
  const int n  = lane & 15;
  const int kb = (lane & 16) ? 16 : 0;
  const _Float16* r = base + n * ld + kb;
  v16h b;
#pragma unroll
  for (int i = 0; i < 8; ++i) { b[2*i] = r[2*i]; b[2*i+1] = r[2*i+1]; }
  return b;
}
__device__ __forceinline__ v8f wmma_f16(v16h a, v16h b, v8f c) {
  return __builtin_amdgcn_wmma_f32_16x16x32_f16(false, a, false, b,
                                                (short)0, c, false, false);
}

__device__ __forceinline__ void cvt4(_Float16* d, float4 v) {
  v4h t = { (_Float16)v.x, (_Float16)v.y, (_Float16)v.z, (_Float16)v.w };
  *(v4h*)d = t;
}

__device__ __forceinline__ int rel_idx(int i, int j) {
  int yi = i / 7, xi = i - yi * 7;
  int yj = j / 7, xj = j - yj * 7;
  return (yi - yj + 6) * 13 + (xi - xj + 6);
}

// ---------------------------------------------------------------------------
// Tensor Data Mover: 1-D contiguous global->LDS copy of NTOK*HD f16 elements,
// with LDS padding of 4 DWORDs after every 16 DWORDs (=> stride 32->40 halfs).
typedef __attribute__((address_space(3))) char lds_char;
__device__ __forceinline__ unsigned lds_off(void* p) {
  return (unsigned)(unsigned long long)(lds_char*)p;
}

__device__ __forceinline__ void tdm_load_qk(unsigned lds_byte, unsigned long long ga) {
  u32x4 g0;
  g0[0] = 1u;                                   // count=1, user descriptor
  g0[1] = lds_byte;                             // lds_addr
  g0[2] = (unsigned)ga;                         // global_addr[31:0]
  g0[3] = (unsigned)((ga >> 32) & 0x01FFFFFFu)  // global_addr[56:32]
        | (2u << 30);                           // type = 2 ("image")
  const unsigned long long dim0 = (unsigned long long)(NTOK * HD_);  // 1568 elems
  union { unsigned long long q[4]; i32x8 v; } g1;
  // data_size=1 (2B) | pad_enable | pad_interval=3 (16 DW) | pad_amount=3 (4 DW)
  g1.q[0] = (1ull << 16) | (1ull << 20) | (3ull << 22) | (3ull << 25)
          | ((dim0 & 0xFFFFull) << 48);         // tensor_dim0[15:0] @ bits 63:48
  g1.q[1] = (dim0 >> 16)                        // tensor_dim0[31:16]
          | (1ull << 16)                        // tensor_dim1 = 1
          | (dim0 << 48);                       // tile_dim0 = 1568
  g1.q[2] = 1ull                                // tile_dim1 = 1 (tile_dim2 = 0)
          | (dim0 << 32);                       // tensor_dim0_stride[31:0]
  g1.q[3] = 0;                                  // stride hi / dim1 stride unused
  i32x4 z4 = {0, 0, 0, 0};
  i32x8 z8 = {0, 0, 0, 0, 0, 0, 0, 0};
  __builtin_amdgcn_tensor_load_to_lds(g0, g1.v, z4, z4, z8, 0);
}

// ---------------------------------------------------------------------------
// Kernel 1: QKV projection, M=100352 N=1152 K=384, 64x64 tiles, double-buffered.
#define TLD 40

__global__ __launch_bounds__(256) void qkv_gemm_kernel(
    const float* __restrict__ x, const float* __restrict__ qkv_w,
    const float* __restrict__ qkv_b, _Float16* __restrict__ qkv_out) {
  __shared__ alignas(16) _Float16 aT[2][64 * TLD];
  __shared__ alignas(16) _Float16 bT[2][64 * TLD];
  const int tid  = threadIdx.x;
  const int lane = tid & 31, wave = tid >> 5;
  const int mBase = blockIdx.x * 64;
  const int nBase = blockIdx.y * 64;
  const int mt = wave >> 1, nh = wave & 1;

  // staging map: thread -> rows r0 and r0+32, 4 consecutive floats at c0
  const int r0 = tid >> 3, c0 = (tid & 7) * 4;
  const float* xr0 = x     + (size_t)(mBase + r0) * C_ + c0;
  const float* wr0 = qkv_w + (size_t)(nBase + r0) * C_ + c0;
  const float* xr1 = xr0 + (size_t)32 * C_;
  const float* wr1 = wr0 + (size_t)32 * C_;

  float4 ra0, ra1, rb0, rb1;
  ra0 = *(const float4*)(xr0); ra1 = *(const float4*)(xr1);
  rb0 = *(const float4*)(wr0); rb1 = *(const float4*)(wr1);
  cvt4(&aT[0][r0 * TLD + c0], ra0); cvt4(&aT[0][(r0 + 32) * TLD + c0], ra1);
  cvt4(&bT[0][r0 * TLD + c0], rb0); cvt4(&bT[0][(r0 + 32) * TLD + c0], rb1);
  __syncthreads();

  v8f acc[2] = {};
  for (int ks = 0; ks < 12; ++ks) {
    const int buf = ks & 1;
    if (ks < 11) {                      // issue next panel loads early
      const int k1 = (ks + 1) * 32;
      ra0 = *(const float4*)(xr0 + k1); ra1 = *(const float4*)(xr1 + k1);
      rb0 = *(const float4*)(wr0 + k1); rb1 = *(const float4*)(wr1 + k1);
      __builtin_prefetch(xr0 + k1 + 32, 0, 1);
      __builtin_prefetch(wr0 + k1 + 32, 0, 1);
    }
    v16h a = load_a16(&aT[buf][mt * 16 * TLD], TLD, lane);
#pragma unroll
    for (int f = 0; f < 2; ++f) {
      v16h b = load_b16(&bT[buf][(nh * 2 + f) * 16 * TLD], TLD, lane);
      acc[f] = wmma_f16(a, b, acc[f]);
    }
    if (ks < 11) {
      const int nb = buf ^ 1;
      cvt4(&aT[nb][r0 * TLD + c0], ra0); cvt4(&aT[nb][(r0 + 32) * TLD + c0], ra1);
      cvt4(&bT[nb][r0 * TLD + c0], rb0); cvt4(&bT[nb][(r0 + 32) * TLD + c0], rb1);
    }
    __syncthreads();
  }
  // scatter to f16 qkv[s][w][h][t][d], q pre-scaled by SCALE_.
  // One division for the whole thread; token index carried incrementally.
  const int gr0 = mBase + mt * 16 + ((lane & 16) ? 8 : 0);
  const int w0 = gr0 / NTOK, t0 = gr0 - w0 * NTOK;
#pragma unroll
  for (int f = 0; f < 2; ++f) {
    const int gc = nBase + (nh * 2 + f) * 16 + (lane & 15);
    const int s = gc / C_, rem = gc - s * C_;
    const int h = rem >> 5, d = rem & 31;
    const float bias = qkv_b[gc];
    const float scl = (s == 0) ? SCALE_ : 1.0f;
    int w = w0, t = t0;
#pragma unroll
    for (int j = 0; j < 8; ++j) {
      float v = (acc[f][j] + bias) * scl;
      size_t o = ((((size_t)s * BW_ + w) * NH_ + h) * NTOK + t) * HD_ + d;
      qkv_out[o] = (_Float16)v;
      if (++t == NTOK) { t = 0; ++w; }
    }
  }
}

// ---------------------------------------------------------------------------
// Kernel 2: attention per (window, head), 49 padded to 64.
#define QLD 40
#define SLD 68
#define PLD 72

__global__ __launch_bounds__(256) void attn_kernel(
    const _Float16* __restrict__ qkv, const float* __restrict__ mask,
    const float* __restrict__ bias_table, _Float16* __restrict__ attn_out) {
  __shared__ alignas(16) _Float16 qs [64 * QLD];
  __shared__ alignas(16) _Float16 kts[64 * QLD];
  __shared__ alignas(16) _Float16 vts[32 * PLD];
  __shared__ alignas(16) float    Ss [64 * SLD];
  __shared__ alignas(16) _Float16 Ps [64 * PLD];
  __shared__ float bias_h[169];

  const int h = blockIdx.x;            // 12
  const int w = blockIdx.y;            // 2048
  const int tid = threadIdx.x;
  const int lane = tid & 31, wave = tid >> 5;

  const size_t qoff = (((size_t)0 * BW_ + w) * NH_ + h) * (NTOK * HD_);
  const size_t koff = (((size_t)1 * BW_ + w) * NH_ + h) * (NTOK * HD_);
  const size_t voff = (((size_t)2 * BW_ + w) * NH_ + h) * (NTOK * HD_);

  // Tensor Data Mover: q and k tiles -> padded LDS (rows 49..63 stay garbage;
  // softmax below forces P=0 for every invalid row/col so it never escapes).
  if (wave == 0) {
    tdm_load_qk(lds_off(qs),  (unsigned long long)(qkv + qoff));
    tdm_load_qk(lds_off(kts), (unsigned long long)(qkv + koff));
  }

  for (int i = tid; i < 169; i += 256) bias_h[i] = bias_table[i * NH_ + h];

  { // v^T staging: 8 strided u16 loads -> one 16B LDS store per thread
    const int d = tid >> 3, j0 = (tid & 7) * 8;
    v8h t;
#pragma unroll
    for (int i = 0; i < 8; ++i) {
      int j = j0 + i;
      t[i] = (j < NTOK) ? qkv[voff + (size_t)j * HD_ + d] : (_Float16)0.f;
    }
    *(v8h*)&vts[d * PLD + j0] = t;
  }
  if (wave == 0) __builtin_amdgcn_s_wait_tensorcnt(0);
  __syncthreads();

  { // S = (q*scale) @ k^T : K=32, one WMMA per 16x16 tile
    const int mt = wave >> 1, nh2 = wave & 1;
    v16h a = load_a16(&qs[mt * 16 * QLD], QLD, lane);
#pragma unroll
    for (int f = 0; f < 2; ++f) {
      int nt = nh2 * 2 + f;
      v16h b = load_b16(&kts[nt * 16 * QLD], QLD, lane);
      v8f c = {};
      c = wmma_f16(a, b, c);
      int col = nt * 16 + (lane & 15);
      int rb  = mt * 16 + ((lane & 16) ? 8 : 0);
#pragma unroll
      for (int j = 0; j < 8; ++j) Ss[(rb + j) * SLD + col] = c[j];
    }
  }
  __syncthreads();

  { // softmax with rel-pos bias + window mask; write P f16 (zeros outside 49x49)
    const int row = tid >> 2, q4 = tid & 3;
    const int c0 = q4 * 16;
    const float* mrow =
        mask + ((size_t)(w & (NWIN - 1)) * NTOK + (row < NTOK ? row : 0)) * NTOK;
    float vals[16];
    float mx = -3.0e38f;
#pragma unroll
    for (int jj = 0; jj < 16; ++jj) {
      int col = c0 + jj;
      float v = -3.0e38f;
      if (row < NTOK && col < NTOK)
        v = Ss[row * SLD + col] + bias_h[rel_idx(row, col)] + mrow[col];
      vals[jj] = v;
      mx = fmaxf(mx, v);
    }
    mx = fmaxf(mx, __shfl_xor(mx, 1, 32));
    mx = fmaxf(mx, __shfl_xor(mx, 2, 32));
    float sum = 0.f;
#pragma unroll
    for (int jj = 0; jj < 16; ++jj) {
      int col = c0 + jj;
      float p = (row < NTOK && col < NTOK) ? __expf(vals[jj] - mx) : 0.f;
      vals[jj] = p;
      sum += p;
    }
    sum += __shfl_xor(sum, 1, 32);
    sum += __shfl_xor(sum, 2, 32);
    float inv = (row < NTOK) ? (1.0f / sum) : 0.f;
#pragma unroll
    for (int jj = 0; jj < 16; ++jj)
      Ps[row * PLD + c0 + jj] = (_Float16)(vals[jj] * inv);
  }
  __syncthreads();

  { // O = P @ V (K=64 -> 2 WMMA steps); store f16 [w][t][h*32+d]
    const int mt = wave >> 1, nt = wave & 1;
    v8f c = {};
#pragma unroll
    for (int ks = 0; ks < 2; ++ks) {
      v16h a = load_a16(&Ps [mt * 16 * PLD + ks * 32], PLD, lane);
      v16h b = load_b16(&vts[nt * 16 * PLD + ks * 32], PLD, lane);
      c = wmma_f16(a, b, c);
    }
    int col = nt * 16 + (lane & 15);
    int rb  = mt * 16 + ((lane & 16) ? 8 : 0);
#pragma unroll
    for (int j = 0; j < 8; ++j) {
      int row = rb + j;
      if (row < NTOK)
        attn_out[((size_t)w * NTOK + row) * C_ + h * HD_ + col] = (_Float16)c[j];
    }
  }
}

// ---------------------------------------------------------------------------
// Kernel 3: output projection, M=100352 N=384 K=384, double-buffered.
__global__ __launch_bounds__(256) void proj_gemm_kernel(
    const _Float16* __restrict__ attn, const float* __restrict__ proj_w,
    const float* __restrict__ proj_b, float* __restrict__ out) {
  __shared__ alignas(16) _Float16 aT[2][64 * TLD];
  __shared__ alignas(16) _Float16 bT[2][64 * TLD];
  const int tid = threadIdx.x, lane = tid & 31, wave = tid >> 5;
  const int mBase = blockIdx.x * 64;
  const int nBase = blockIdx.y * 64;
  const int mt = wave >> 1, nh = wave & 1;

  // A (f16): one 16B load/store per thread; B (f32): two float4 per thread
  const int ar = tid >> 2, ac = (tid & 3) * 8;
  const int br = tid >> 3, bc = (tid & 7) * 4;
  const _Float16* arow = attn   + (size_t)(mBase + ar) * C_ + ac;
  const float*    wr0  = proj_w + (size_t)(nBase + br) * C_ + bc;
  const float*    wr1  = wr0 + (size_t)32 * C_;

  v8h    raa = *(const v8h*)(arow);
  float4 rb0 = *(const float4*)(wr0);
  float4 rb1 = *(const float4*)(wr1);
  *(v8h*)&aT[0][ar * TLD + ac] = raa;
  cvt4(&bT[0][br * TLD + bc], rb0);
  cvt4(&bT[0][(br + 32) * TLD + bc], rb1);
  __syncthreads();

  v8f acc[2] = {};
  for (int ks = 0; ks < 12; ++ks) {
    const int buf = ks & 1;
    if (ks < 11) {
      const int k1 = (ks + 1) * 32;
      raa = *(const v8h*)(arow + k1);
      rb0 = *(const float4*)(wr0 + k1);
      rb1 = *(const float4*)(wr1 + k1);
      __builtin_prefetch(arow + k1 + 32, 0, 1);
    }
    v16h a = load_a16(&aT[buf][mt * 16 * TLD], TLD, lane);
#pragma unroll
    for (int f = 0; f < 2; ++f) {
      v16h b = load_b16(&bT[buf][(nh * 2 + f) * 16 * TLD], TLD, lane);
      acc[f] = wmma_f16(a, b, acc[f]);
    }
    if (ks < 11) {
      const int nb = buf ^ 1;
      *(v8h*)&aT[nb][ar * TLD + ac] = raa;
      cvt4(&bT[nb][br * TLD + bc], rb0);
      cvt4(&bT[nb][(br + 32) * TLD + bc], rb1);
    }
    __syncthreads();
  }
#pragma unroll
  for (int f = 0; f < 2; ++f) {
    int gc = nBase + (nh * 2 + f) * 16 + (lane & 15);
    float bias = proj_b[gc];
#pragma unroll
    for (int j = 0; j < 8; ++j) {
      int gr = mBase + mt * 16 + j + ((lane & 16) ? 8 : 0);
      out[(size_t)gr * C_ + gc] = acc[f][j] + bias;
    }
  }
}

// ---------------------------------------------------------------------------
extern "C" void kernel_launch(void* const* d_in, const int* in_sizes, int n_in,
                              void* d_out, int out_size, void* d_ws, size_t ws_size,
                              hipStream_t stream) {
  const float* x        = (const float*)d_in[0];
  const float* mask     = (const float*)d_in[1];
  const float* qkv_w    = (const float*)d_in[2];
  const float* qkv_b    = (const float*)d_in[3];
  const float* proj_w   = (const float*)d_in[4];
  const float* proj_b   = (const float*)d_in[5];
  const float* bias_tab = (const float*)d_in[6];
  float* out = (float*)d_out;

  _Float16* qkv_ws  = (_Float16*)d_ws;
  _Float16* attn_ws = qkv_ws + (size_t)3 * BW_ * NH_ * NTOK * HD_;

  qkv_gemm_kernel <<<dim3(1568, 18), 256, 0, stream>>>(x, qkv_w, qkv_b, qkv_ws);
  attn_kernel     <<<dim3(NH_, BW_), 256, 0, stream>>>(qkv_ws, mask, bias_tab, attn_ws);
  proj_gemm_kernel<<<dim3(1568, 6), 256, 0, stream>>>(attn_ws, proj_w, proj_b, out);
}
